// MinkUNet_86947317940493
// MI455X (gfx1250) — matrix operand order repaired and checked by reading, not compiled, not run on hardware
//
#include <hip/hip_runtime.h>

// ---------------------------------------------------------------------------
// CDNA5 (gfx1250) fused LocalFeatureAggregation.
//   N=50000 pts, K=16 neighbors, INC=256, MID=64, 2*MID=128, OUTC=256.
//   bf16 WMMA (v_wmma_f32_16x16x32_bf16) for all GEMMs; pos-MLP (K=10) in VALU.
//   Weights pre-converted to bf16 in ws; neighbor-feature gather uses
//   global_load_async_to_lds_b128 (ASYNCcnt) straight into the LDS stack.
// ---------------------------------------------------------------------------

typedef __attribute__((ext_vector_type(16))) __bf16 v16bf;
typedef __attribute__((ext_vector_type(8)))  float  v8f;

#define WMMA_BF16(acc, a, b) \
    __builtin_amdgcn_wmma_f32_16x16x32_bf16(false, (a), false, (b), (short)0, (acc), false, false)

// 16-bit A-matrix 16x32 striping (matches B):
//  lane L: row M = L&15, half g = L>>4.
//  element e : K = 16g + e (contiguous)
__device__ __forceinline__ int kA(int e, int g) {
    return 16 * g + e;
}
// B-matrix 32x16 order used here (same as A, not AMD's table order): lane L col N = L&15 ; VGPR v: K = 2v + 16g (+1)
// -> per lane, the 16 elements are the CONTIGUOUS range [k0+16g, k0+16g+16).
__device__ __forceinline__ v16bf loadB_bf(const __bf16* __restrict__ W, int Kdim, int n0, int k0, int lane) {
    int n = n0 + (lane & 15), g = lane >> 4;
    const __bf16* p = W + (size_t)n * Kdim + k0 + 16 * g;
    v16bf b;
#pragma unroll
    for (int e = 0; e < 16; ++e) b[e] = p[e];
    return b;
}
// A fragment from a row-major bf16 LDS tile
__device__ __forceinline__ v16bf loadA_bf(const __bf16* s, int stride, int row0, int k0, int lane) {
    int m = lane & 15, g = lane >> 4;
    v16bf a;
#pragma unroll
    for (int e = 0; e < 16; ++e) a[e] = s[(row0 + m) * stride + k0 + kA(e, g)];
    return a;
}
// A fragment from a row-major f32 LDS tile (convert on the fly)
__device__ __forceinline__ v16bf loadA_f32(const float* s, int stride, int k0, int lane) {
    int m = lane & 15, g = lane >> 4;
    v16bf a;
#pragma unroll
    for (int e = 0; e < 16; ++e) a[e] = (__bf16)s[m * stride + k0 + kA(e, g)];
    return a;
}

// ---------------------------------------------------------------------------
// Prep: convert all weight matrices f32 -> bf16 into workspace (L2-resident).
// Segments: w_in [64*256) | w_skip [256*256) | w_att [128*128) | w_out [256*128)
// ---------------------------------------------------------------------------
__global__ void __launch_bounds__(256) lfa_cvt_weights(
    const float* __restrict__ w_in, const float* __restrict__ w_skip,
    const float* __restrict__ w_att, const float* __restrict__ w_out,
    __bf16* __restrict__ dst) {
    int i = blockIdx.x * 256 + threadIdx.x;
    const int S0 = 64 * 256, S1 = S0 + 256 * 256, S2 = S1 + 128 * 128, S3 = S2 + 256 * 128;
    if (i < S0)      dst[i] = (__bf16)w_in[i];
    else if (i < S1) dst[i] = (__bf16)w_skip[i - S0];
    else if (i < S2) dst[i] = (__bf16)w_att[i - S1];
    else if (i < S3) dst[i] = (__bf16)w_out[i - S2];
}

// ---------------------------------------------------------------------------
// Kernel 1: x1[N,64] (bf16) = x[N,256] @ w_in.T  (global, for neighbor gather)
// 128 threads (4 waves) per 16-row tile; each wave owns one 16-col N-tile.
// ---------------------------------------------------------------------------
__global__ void __launch_bounds__(128) lfa_x1_kernel(
    const float* __restrict__ x, const __bf16* __restrict__ wb_in,
    __bf16* __restrict__ x1b, int N) {
    __shared__ __bf16 sX[16 * 256];
    int base = blockIdx.x * 16;
    for (int i = threadIdx.x; i < 16 * 256; i += 128) {
        int r = base + (i >> 8);
        if (r >= N) r = N - 1;
        sX[i] = (__bf16)x[r * 256 + (i & 255)];
    }
    __syncthreads();
    int wave = threadIdx.x >> 5, lane = threadIdx.x & 31;
    v8f acc = {};
#pragma unroll
    for (int ks = 0; ks < 8; ++ks) {
        v16bf a = loadA_bf(sX, 256, 0, 32 * ks, lane);
        v16bf b = loadB_bf(wb_in, 256, 16 * wave, 32 * ks, lane);
        acc = WMMA_BF16(acc, a, b);
    }
    int g = lane >> 4, c = 16 * wave + (lane & 15);
#pragma unroll
    for (int r = 0; r < 8; ++r) {
        int m = base + r + 8 * g;
        if (m < N) x1b[(size_t)m * 64 + c] = (__bf16)acc[r];
    }
}

// ---------------------------------------------------------------------------
// Kernel 2: fully fused LFA for a 16-point tile (256 threads = 8 waves).
//   Attention GEMM M-tiles (16 rows) == one point's 16 neighbors -> softmax is
//   an in-tile reduction (8 acc VGPRs + shfl_xor(16)).
// ---------------------------------------------------------------------------
__global__ void __launch_bounds__(256) lfa_fused_kernel(
    const float* __restrict__ x,   const int* __restrict__ pos,
    const int* __restrict__ idx,   const float* __restrict__ dist,
    const __bf16* __restrict__ x1b,
    const __bf16* __restrict__ wb_skip, const __bf16* __restrict__ wb_att,
    const __bf16* __restrict__ wb_out,
    const float* __restrict__ w_pos, const float* __restrict__ b_pos,
    const float* __restrict__ g_pos, const float* __restrict__ be_pos,
    const float* __restrict__ m_pos, const float* __restrict__ v_pos,
    const float* __restrict__ b_att,
    const float* __restrict__ g_att, const float* __restrict__ be_att,
    const float* __restrict__ m_att, const float* __restrict__ v_att,
    const float* __restrict__ b_out,
    const float* __restrict__ g_out, const float* __restrict__ be_out,
    const float* __restrict__ m_out, const float* __restrict__ v_out,
    float* __restrict__ out, int N) {
    extern __shared__ char smemRaw[];
    float*  sAgg   = (float*)smemRaw;            // 16*128 f32
    float*  sWpos  = sAgg + 16 * 128;            // 64*10
    float*  sPosA  = sWpos + 640;                // 64  folded BN scale
    float*  sPosB  = sPosA + 64;                 // 64  folded BN shift
    float*  sAttA  = sPosB + 64;                 // 128
    float*  sAttB  = sAttA + 128;                // 128
    float*  sOutA  = sAttB + 128;                // 256
    float*  sOutB  = sOutA + 256;                // 256
    __bf16* sX     = (__bf16*)(sOutB + 256);     // 16*256
    __bf16* sStack = sX + 16 * 256;              // 256*128  (fxyz | ffeats)

    const int t = threadIdx.x;
    const int base = blockIdx.x * 16;

    // stage pos-MLP weights + folded BN params (bn(y+b) = y*sc + (be + (b-m)*sc))
    for (int i = t; i < 640; i += 256) sWpos[i] = w_pos[i];
    if (t < 64) {
        float sc = g_pos[t] * rsqrtf(v_pos[t] + 1e-5f);
        sPosA[t] = sc;
        sPosB[t] = be_pos[t] + (b_pos[t] - m_pos[t]) * sc;
    }
    if (t < 128) {
        float sc = g_att[t] * rsqrtf(v_att[t] + 1e-5f);
        sAttA[t] = sc;
        sAttB[t] = be_att[t] + (b_att[t] - m_att[t]) * sc;
    }
    {
        float sc = g_out[t] * rsqrtf(v_out[t] + 1e-5f);
        sOutA[t] = sc;
        sOutB[t] = be_out[t] + (b_out[t] - m_out[t]) * sc;
    }
    // stage x tile (coalesced)
    for (int i = t; i < 16 * 256; i += 256) {
        int r = base + (i >> 8);
        if (r >= N) r = N - 1;
        sX[i] = (__bf16)x[r * 256 + (i & 255)];
    }

    // Neighbor-feature gather: async DMA 128B of x1b[nb] directly into LDS
    // (gfx1250 GLOBAL_LOAD_ASYNC_TO_LDS_B128, tracked by ASYNCcnt; x1b is
    //  only 6.4MB -> fully L2-resident, gathers never touch HBM).
    int pl = t >> 4, kk = t & 15;
    int pidx = base + pl; if (pidx >= N) pidx = N - 1;
    const int nb = idx[pidx * 16 + kk];
    {
        unsigned long long ga = (unsigned long long)(uintptr_t)(x1b + (size_t)nb * 64);
        unsigned lds = (unsigned)(uintptr_t)(sStack + t * 128 + 64);
        asm volatile("global_load_async_to_lds_b128 %0, %1, off\n\t"
                     "global_load_async_to_lds_b128 %0, %1, off offset:16\n\t"
                     "global_load_async_to_lds_b128 %0, %1, off offset:32\n\t"
                     "global_load_async_to_lds_b128 %0, %1, off offset:48\n\t"
                     "global_load_async_to_lds_b128 %0, %1, off offset:64\n\t"
                     "global_load_async_to_lds_b128 %0, %1, off offset:80\n\t"
                     "global_load_async_to_lds_b128 %0, %1, off offset:96\n\t"
                     "global_load_async_to_lds_b128 %0, %1, off offset:112"
                     :: "v"(lds), "v"(ga) : "memory");
    }

    // pos-MLP half of the stack row: [fxyz(64) | ffeats(64)]
    {
        float d  = dist[pidx * 16 + kk];
        float cx = (float)pos[pidx * 4 + 0], cy = (float)pos[pidx * 4 + 1], cz = (float)pos[pidx * 4 + 2];
        float nx = (float)pos[nb * 4 + 0],   ny = (float)pos[nb * 4 + 1],   nz = (float)pos[nb * 4 + 2];
        float pe[10] = {d, cx - nx, cy - ny, cz - nz, cx, cy, cz, nx, ny, nz};
#pragma unroll 4
        for (int c = 0; c < 64; ++c) {
            float s = 0.f;
#pragma unroll
            for (int j = 0; j < 10; ++j) s += pe[j] * sWpos[c * 10 + j];
            s = fmaxf(s * sPosA[c] + sPosB[c], 0.f);
            sStack[t * 128 + c] = (__bf16)s;
        }
    }
    asm volatile("s_wait_asynccnt 0" ::: "memory");   // ffeats DMA complete
    __syncthreads();

    const int wave = t >> 5, lane = t & 31, g = lane >> 4;

    // Attention GEMM [256,128]x[128,128] + BN + relu + softmax(K) + pooling.
    // 128 (mt,nt) tile-pairs over 8 waves; mt == point index in tile.
    for (int q = wave * 16; q < wave * 16 + 16; ++q) {
        int mt = q >> 3, nt = q & 7;
        v8f acc = {};
#pragma unroll
        for (int ks = 0; ks < 4; ++ks) {
            v16bf a = loadA_bf(sStack, 128, 16 * mt, 32 * ks, lane);
            v16bf b = loadB_bf(wb_att, 128, 16 * nt, 32 * ks, lane);
            acc = WMMA_BF16(acc, a, b);
        }
        int c = 16 * nt + (lane & 15);
        float sc = sAttA[c], sh = sAttB[c];
        float f[8]; float mx = -3.4e38f;
#pragma unroll
        for (int r = 0; r < 8; ++r) { f[r] = fmaxf(acc[r] * sc + sh, 0.f); mx = fmaxf(mx, f[r]); }
        mx = fmaxf(mx, __shfl_xor(mx, 16, 32));          // combine M halves (k = r+8g)
        float sum = 0.f;
#pragma unroll
        for (int r = 0; r < 8; ++r) { f[r] = __expf(f[r] - mx); sum += f[r]; }
        sum += __shfl_xor(sum, 16, 32);
        float inv = 1.f / sum;
        float part = 0.f;
#pragma unroll
        for (int r = 0; r < 8; ++r)
            part += f[r] * inv * (float)sStack[(16 * mt + r + 8 * g) * 128 + c];
        part += __shfl_xor(part, 16, 32);
        if (lane < 16) sAgg[mt * 128 + c] = part;
    }
    __syncthreads();

    // out = leaky_relu( x@w_skip.T + relu(bn(agg@w_out.T + b_out)) ), per 16-col tile
    for (int nt = 2 * wave; nt < 2 * wave + 2; ++nt) {
        v8f accS = {};
#pragma unroll
        for (int ks = 0; ks < 8; ++ks) {
            v16bf a = loadA_bf(sX, 256, 0, 32 * ks, lane);
            v16bf b = loadB_bf(wb_skip, 256, 16 * nt, 32 * ks, lane);
            accS = WMMA_BF16(accS, a, b);
        }
        v8f accL = {};
#pragma unroll
        for (int ks = 0; ks < 4; ++ks) {
            v16bf a = loadA_f32(sAgg, 128, 32 * ks, lane);
            v16bf b = loadB_bf(wb_out, 128, 16 * nt, 32 * ks, lane);
            accL = WMMA_BF16(accL, a, b);
        }
        int c = 16 * nt + (lane & 15);
        float sc = sOutA[c], sh = sOutB[c];
#pragma unroll
        for (int r = 0; r < 8; ++r) {
            int m = base + r + 8 * g;
            if (m < N) {
                float l = fmaxf(accL[r] * sc + sh, 0.f);
                float o = accS[r] + l;
                out[(size_t)m * 256 + c] = (o > 0.f) ? o : 0.1f * o;
            }
        }
    }
}

// ---------------------------------------------------------------------------
extern "C" void kernel_launch(void* const* d_in, const int* in_sizes, int n_in,
                              void* d_out, int out_size, void* d_ws, size_t ws_size,
                              hipStream_t stream) {
    (void)n_in; (void)out_size; (void)ws_size;
    const float* x      = (const float*)d_in[0];
    const int*   pos    = (const int*)d_in[1];
    const int*   idx    = (const int*)d_in[2];
    const float* dist   = (const float*)d_in[3];
    const float* w_in   = (const float*)d_in[4];
    const float* w_skip = (const float*)d_in[5];
    const float* w_pos  = (const float*)d_in[6];
    const float* b_pos  = (const float*)d_in[7];
    const float* g_pos  = (const float*)d_in[8];
    const float* be_pos = (const float*)d_in[9];
    const float* m_pos  = (const float*)d_in[10];
    const float* v_pos  = (const float*)d_in[11];
    const float* b_att  = (const float*)d_in[13];
    const float* g_att  = (const float*)d_in[14];
    const float* be_att = (const float*)d_in[15];
    const float* m_att  = (const float*)d_in[16];
    const float* v_att  = (const float*)d_in[17];
    const float* w_att  = (const float*)d_in[12];
    const float* w_out  = (const float*)d_in[18];
    const float* b_out  = (const float*)d_in[19];
    const float* g_out  = (const float*)d_in[20];
    const float* be_out = (const float*)d_in[21];
    const float* m_out  = (const float*)d_in[22];
    const float* v_out  = (const float*)d_in[23];

    const int N = in_sizes[0] / 256;               // x is [N, 256]
    const int nblk = (N + 15) / 16;

    // workspace layout: x1 (bf16, N*64) | bf16 weights (w_in|w_skip|w_att|w_out)
    __bf16* x1b     = (__bf16*)d_ws;
    __bf16* wb_in   = x1b + (size_t)N * 64;
    __bf16* wb_skip = wb_in + 64 * 256;
    __bf16* wb_att  = wb_skip + 256 * 256;
    __bf16* wb_out  = wb_att + 128 * 128;

    const int totW = 64 * 256 + 256 * 256 + 128 * 128 + 256 * 128;   // 131072
    lfa_cvt_weights<<<(totW + 255) / 256, 256, 0, stream>>>(w_in, w_skip, w_att, w_out, wb_in);
    lfa_x1_kernel<<<nblk, 128, 0, stream>>>(x, wb_in, x1b, N);

    // dynamic LDS: f32 params/agg + bf16 x-tile + bf16 stack  (~88 KB of 320 KB WGP LDS)
    const size_t smem = (size_t)(16 * 128 + 640 + 64 + 64 + 128 + 128 + 256 + 256) * sizeof(float)
                      + (size_t)(16 * 256 + 256 * 128) * sizeof(__bf16);
    lfa_fused_kernel<<<nblk, 256, smem, stream>>>(
        x, pos, idx, dist, x1b, wb_skip, wb_att, wb_out,
        w_pos, b_pos, g_pos, be_pos, m_pos, v_pos,
        b_att, g_att, be_att, m_att, v_att,
        b_out, g_out, be_out, m_out, v_out,
        (float*)d_out, N);
}
